// ConvexLinearAttention_88072599372589
// MI455X (gfx1250) — compile-verified, matrix operand order checked by reading, not run on hardware
//
#include <hip/hip_runtime.h>
#include <hip/hip_bf16.h>

// ---------------------------------------------------------------------------
// ConvexLinearAttention for MI455X (gfx1250, wave32, WMMA).
// Linear-attention factorization: attended = (qf @ KV) / max(qf . ksum, eps),
// KV = sum_j kf_j^T v_j. All heavy GEMMs use v_wmma_f32_16x16x32_f16.
// GEMM tile staging uses GLOBAL_LOAD_ASYNC_TO_LDS_B128 (ASYNCcnt) with double
// buffering when the toolchain exposes the builtin; ds_store fallback else.
// ---------------------------------------------------------------------------

typedef __attribute__((ext_vector_type(16))) _Float16 v16h;
typedef __attribute__((ext_vector_type(8)))  _Float16 v8h;
typedef __attribute__((ext_vector_type(4)))  _Float16 v4h;
typedef __attribute__((ext_vector_type(8)))  float    v8f;
typedef __attribute__((ext_vector_type(4)))  float    v4f;
typedef __attribute__((ext_vector_type(4)))  int      v4i;

#if defined(__has_builtin)
#  if __has_builtin(__builtin_amdgcn_global_load_async_to_lds_b128) && \
      __has_builtin(__builtin_amdgcn_s_wait_asynccnt)
#    define HAVE_ASYNC_LDS 1
#  endif
#endif

#define WMMA16(a, b, c) \
  __builtin_amdgcn_wmma_f32_16x16x32_f16(false, (a), false, (b), (short)0, (c), false, false)

// Load one 16x32 f16 WMMA fragment from a row-major matrix with leading dim
// `ld` (in elements). Per ISA 7.12.2: lane L holds row L%16; lanes 0-15 carry
// K=[0..7]+[16..23], lanes 16-31 carry K=[8..15]+[24..31]. Two b128 loads.
__device__ __forceinline__ v16h frag_ld(const _Float16* __restrict__ p, int ld, int lane) {
  const int r  = lane & 15;
  const int hf = lane >> 4;
  const _Float16* q = p + r * ld + hf * 8;
  v8h lo = *(const v8h*)(q);
  v8h hi = *(const v8h*)(q + 16);
  return __builtin_shufflevector(lo, hi, 0,1,2,3,4,5,6,7,8,9,10,11,12,13,14,15);
}

#ifdef HAVE_ASYNC_LDS
// Builtin signature (from hipcc diagnostic): param0 is AS1 (global, "__device__")
// pointer to int4, non-const; param1 is the LDS-side pointer; args 2/3 are
// immediate offset / cache-policy.
typedef __attribute__((address_space(1))) v4i* gv4i_p;
typedef __attribute__((address_space(3))) v4i* lv4i_p;

__device__ __forceinline__ void async_cp16(const _Float16* g, _Float16* l) {
  __builtin_amdgcn_global_load_async_to_lds_b128((gv4i_p)g, (lv4i_p)l, 0, 0);
}
#endif

// ---------------------------------------------------------------------------
// f32 -> f16 conversion, 4 elements/thread (b128 in, b64 out)
// ---------------------------------------------------------------------------
__global__ __launch_bounds__(256) void cvt_f32_f16(const float* __restrict__ in,
                                                   _Float16* __restrict__ out, int n4) {
  int i = blockIdx.x * 256 + threadIdx.x;
  if (i < n4) {
    v4f f = *(const v4f*)&in[i * 4];
    v4h o = {(_Float16)f.x, (_Float16)f.y, (_Float16)f.z, (_Float16)f.w};
    *(v4h*)&out[i * 4] = o;
  }
}

// ---------------------------------------------------------------------------
// NT GEMM: C[m,n] = sum_k A[m,k] * B[n,k] + bias[n], f16 in / f32 accum.
// 128x128 workgroup tile, 8 waves in a 2x4 grid, each wave owns 64x32
// (4x2 WMMA accumulators). K staged through LDS in chunks of 32.
// mode 0: write f32 row-major (final projection -> d_out)
// mode 1: elu+1, scatter f16 to q_feat  [B,H,S,D]
// mode 2: elu+1, scatter f16 to k_featT [B,H,D,S]
// mode 3: no act, scatter f16 to vT     [B,H,D,S]
// ---------------------------------------------------------------------------
#define LDT 40  // padded LDS k-stride (bytes/row = 80, multiple of 16)

__global__ __launch_bounds__(256)
void gemm_nt_f16(const _Float16* __restrict__ A,   // [M,K]
                 const _Float16* __restrict__ Bm,  // [N,K]
                 const float* __restrict__ bias,   // [N]
                 int M, int N, int K, int mode,
                 float* __restrict__ outF,
                 _Float16* __restrict__ outH) {
  const int tid  = threadIdx.x;
  const int lane = tid & 31;
  const int wave = tid >> 5;  // 0..7
  const int wr   = wave >> 2; // 0..1
  const int wc   = wave & 3;  // 0..3
  const int m0   = blockIdx.x * 128;
  const int n0   = blockIdx.y * 128;

  v8f acc[4][2] = {};

#ifdef HAVE_ASYNC_LDS
  // ------------------ async double-buffered pipeline ------------------
  __shared__ _Float16 As[2][128 * LDT];
  __shared__ _Float16 Bs[2][128 * LDT];

  auto issue_tile = [&](int k0, int buf) {
#pragma unroll
    for (int it = 0; it < 2; ++it) {
      int idx = it * 256 + tid;
      int r = idx >> 2;
      int c = (idx & 3) * 8;
      async_cp16(&A[(size_t)(m0 + r) * K + k0 + c], &As[buf][r * LDT + c]);
      async_cp16(&Bm[(size_t)(n0 + r) * K + k0 + c], &Bs[buf][r * LDT + c]);
    }
  };
  auto compute_tile = [&](const _Float16* Ab, const _Float16* Bb) {
    v16h bfrag[2];
#pragma unroll
    for (int j = 0; j < 2; ++j)
      bfrag[j] = frag_ld(&Bb[(wc * 32 + j * 16) * LDT], LDT, lane);
#pragma unroll
    for (int i = 0; i < 4; ++i) {
      v16h afrag = frag_ld(&Ab[(wr * 64 + i * 16) * LDT], LDT, lane);
#pragma unroll
      for (int j = 0; j < 2; ++j)
        acc[i][j] = WMMA16(afrag, bfrag[j], acc[i][j]);
    }
  };

  const int nk = K >> 5;
  issue_tile(0, 0);
  for (int ki = 0; ki < nk; ++ki) {
    const int cur = ki & 1;
    __builtin_amdgcn_s_wait_asynccnt(0);  // this wave's tile `cur` landed
    __syncthreads();                      // all waves landed; prev compute done
    if (ki + 1 < nk) issue_tile((ki + 1) << 5, cur ^ 1);  // overlap with WMMA
    compute_tile(As[cur], Bs[cur]);
  }
#else
  // ------------------ fallback: VGPR-staged tile loads ------------------
  __shared__ _Float16 As1[128 * LDT];
  __shared__ _Float16 Bs1[128 * LDT];
  for (int k0 = 0; k0 < K; k0 += 32) {
#pragma unroll
    for (int it = 0; it < 2; ++it) {
      int idx = it * 256 + tid;
      int r = idx >> 2;
      int c = (idx & 3) * 8;
      *(v8h*)&As1[r * LDT + c] = *(const v8h*)&A[(size_t)(m0 + r) * K + k0 + c];
      *(v8h*)&Bs1[r * LDT + c] = *(const v8h*)&Bm[(size_t)(n0 + r) * K + k0 + c];
    }
    __syncthreads();
    v16h bfrag[2];
#pragma unroll
    for (int j = 0; j < 2; ++j)
      bfrag[j] = frag_ld(&Bs1[(wc * 32 + j * 16) * LDT], LDT, lane);
#pragma unroll
    for (int i = 0; i < 4; ++i) {
      v16h afrag = frag_ld(&As1[(wr * 64 + i * 16) * LDT], LDT, lane);
#pragma unroll
      for (int j = 0; j < 2; ++j)
        acc[i][j] = WMMA16(afrag, bfrag[j], acc[i][j]);
    }
    __syncthreads();
  }
#endif

  // Epilogue. C layout (ISA 7.12.2): VGPR e: lanes 0-15 -> M=e, 16-31 -> M=e+8.
  const int lr = lane & 15;
  const int lh = lane >> 4;
#pragma unroll
  for (int i = 0; i < 4; ++i) {
#pragma unroll
    for (int j = 0; j < 2; ++j) {
      const int tm = m0 + wr * 64 + i * 16;
      const int tn = n0 + wc * 32 + j * 16;
#pragma unroll
      for (int e = 0; e < 8; ++e) {
        int row = tm + e + 8 * lh;
        int col = tn + lr;
        float val = acc[i][j][e] + bias[col];
        if (mode == 0) {
          outF[(size_t)row * N + col] = val;
        } else {
          if (mode != 3) val = (val > 0.0f) ? (val + 1.0f) : __expf(val);  // elu(x)+1
          int b = row >> 11, s = row & 2047;   // S = 2048
          int hh = col >> 6, d = col & 63;     // Dh = 64
          if (mode == 1)
            outH[((size_t)(b * 16 + hh) * 2048 + s) * 64 + d] = (_Float16)val;
          else
            outH[((size_t)(b * 16 + hh) * 64 + d) * 2048 + s] = (_Float16)val;
        }
      }
    }
  }
}

// ---------------------------------------------------------------------------
// Stage 2: per (b,h): KV[d,e] = sum_s kfT[d,s] * vT[e,s]; ksum[d] = sum_s kfT[d,s].
// One workgroup per head; 8 waves cover the 4x4 grid of 16x16 tiles
// (2 tiles per wave) over K = 2048. Writes KV transposed (KVT[e][d], f16) so
// stage 3 can load it as a plain NT B-operand.
// ---------------------------------------------------------------------------
__global__ __launch_bounds__(256)
void kv_kernel(const _Float16* __restrict__ kft,  // [32][64][2048]
               const _Float16* __restrict__ vt,   // [32][64][2048]
               _Float16* __restrict__ kvt,        // [32][64][64]  (e-major)
               float* __restrict__ ksum) {        // [32][64]
  const int bh = blockIdx.x;
  const _Float16* Kp = kft + (size_t)bh * 64 * 2048;
  const _Float16* Vp = vt  + (size_t)bh * 64 * 2048;

  const int tid  = threadIdx.x;
  const int lane = tid & 31;
  const int wave = tid >> 5;
  const int tr   = wave >> 1;        // d tile 0..3
  const int tcb  = (wave & 1) * 2;   // e tile base 0 or 2

  v8f acc[2] = {};
  for (int k0 = 0; k0 < 2048; k0 += 32) {
    v16h a = frag_ld(Kp + (size_t)(tr * 16) * 2048 + k0, 2048, lane);
#pragma unroll
    for (int j = 0; j < 2; ++j) {
      v16h b = frag_ld(Vp + (size_t)((tcb + j) * 16) * 2048 + k0, 2048, lane);
      acc[j] = WMMA16(a, b, acc[j]);
    }
  }
  const int lr = lane & 15, lh = lane >> 4;
#pragma unroll
  for (int j = 0; j < 2; ++j) {
#pragma unroll
    for (int e8 = 0; e8 < 8; ++e8) {
      int d = tr * 16 + e8 + 8 * lh;
      int e = (tcb + j) * 16 + lr;
      kvt[(size_t)bh * 4096 + e * 64 + d] = (_Float16)acc[j][e8];
    }
  }

  // ksum reduction: thread t sums kfT[d = t&63] over s-slice t>>6 (v8h loads).
  __shared__ float red[256];
  const int d = tid & 63, sl = tid >> 6;
  float s = 0.0f;
  const _Float16* kr = Kp + (size_t)d * 2048 + sl * 512;
  for (int i = 0; i < 512; i += 8) {
    v8h v = *(const v8h*)&kr[i];
#pragma unroll
    for (int u = 0; u < 8; ++u) s += (float)v[u];
  }
  red[tid] = s;
  __syncthreads();
  if (tid < 64)
    ksum[bh * 64 + tid] = red[tid] + red[tid + 64] + red[tid + 128] + red[tid + 192];
}

// ---------------------------------------------------------------------------
// Stage 3: attended[s,e] = (qf[s,:] @ KV[:,e]) / max(qf[s,:].ksum, eps).
// Grid: (S/128, B*H). Wave w owns rows [w*16, w*16+16), all 4 e-tiles. K=64.
// Writes merged [B,S,H*Dh] in f16 (A-operand of the output projection).
// ---------------------------------------------------------------------------
__global__ __launch_bounds__(256)
void attn_kernel(const _Float16* __restrict__ qf,   // [32][2048][64]
                 const _Float16* __restrict__ kvt,  // [32][64][64] (e-major)
                 const float* __restrict__ ksum,    // [32][64]
                 _Float16* __restrict__ merged) {   // [4096][1024]
  const int bh = blockIdx.y;
  const int b = bh >> 4, h = bh & 15;
  const int m0 = blockIdx.x * 128;
  const _Float16* Q  = qf  + (size_t)bh * 2048 * 64 + (size_t)m0 * 64;
  const _Float16* KV = kvt + (size_t)bh * 4096;

  __shared__ float rden[128];
  __shared__ float ks[64];
  const int tid = threadIdx.x;
  if (tid < 64) ks[tid] = ksum[bh * 64 + tid];
  __syncthreads();
  if (tid < 128) {
    float s = 0.0f;
    const _Float16* qr = Q + (size_t)tid * 64;
#pragma unroll
    for (int d = 0; d < 64; ++d) s += (float)qr[d] * ks[d];
    rden[tid] = 1.0f / fmaxf(s, 1e-6f);
  }
  __syncthreads();

  const int lane = tid & 31;
  const int wave = tid >> 5;
  v8f acc[4] = {};
#pragma unroll
  for (int k0 = 0; k0 < 64; k0 += 32) {
    v16h a = frag_ld(Q + (size_t)(wave * 16) * 64 + k0, 64, lane);
#pragma unroll
    for (int j = 0; j < 4; ++j) {
      v16h bfr = frag_ld(KV + (size_t)(j * 16) * 64 + k0, 64, lane);
      acc[j] = WMMA16(a, bfr, acc[j]);
    }
  }
  const int lr = lane & 15, lh = lane >> 4;
#pragma unroll
  for (int j = 0; j < 4; ++j) {
#pragma unroll
    for (int e8 = 0; e8 < 8; ++e8) {
      int rl = wave * 16 + e8 + 8 * lh;
      int e  = j * 16 + lr;
      float v = acc[j][e8] * rden[rl];
      int s = m0 + rl;
      merged[((size_t)(b * 2048 + s)) * 1024 + h * 64 + e] = (_Float16)v;
    }
  }
}

// ---------------------------------------------------------------------------
// Host-side launcher
// ---------------------------------------------------------------------------
extern "C" void kernel_launch(void* const* d_in, const int* in_sizes, int n_in,
                              void* d_out, int out_size, void* d_ws, size_t ws_size,
                              hipStream_t stream) {
  (void)in_sizes; (void)n_in; (void)out_size; (void)ws_size;
  const float* x  = (const float*)d_in[0];
  const float* wq = (const float*)d_in[1];
  const float* bq = (const float*)d_in[2];
  const float* wk = (const float*)d_in[3];
  const float* bk = (const float*)d_in[4];
  const float* wv = (const float*)d_in[5];
  const float* bv = (const float*)d_in[6];
  const float* wo = (const float*)d_in[7];
  const float* bo = (const float*)d_in[8];
  float* out = (float*)d_out;

  const int M = 4096;      // B*S
  const int DM = 1024;     // d_model (N and K of the projections)
  const int NTOK = M * DM; // 4,194,304
  const int NW   = DM * DM;

  size_t off = 0;
  char* base = (char*)d_ws;
  auto carve = [&](size_t bytes) {
    void* p = base + off;
    off += (bytes + 255) & ~(size_t)255;
    return p;
  };
  _Float16* xh     = (_Float16*)carve((size_t)NTOK * 2);
  _Float16* wqh    = (_Float16*)carve((size_t)NW * 2);
  _Float16* wkh    = (_Float16*)carve((size_t)NW * 2);
  _Float16* wvh    = (_Float16*)carve((size_t)NW * 2);
  _Float16* woh    = (_Float16*)carve((size_t)NW * 2);
  _Float16* qf     = (_Float16*)carve((size_t)NTOK * 2);  // [B,H,S,D]
  _Float16* kft    = (_Float16*)carve((size_t)NTOK * 2);  // [B,H,D,S]
  _Float16* vt     = (_Float16*)carve((size_t)NTOK * 2);  // [B,H,D,S]
  _Float16* merged = (_Float16*)carve((size_t)NTOK * 2);  // [B*S, H*D]
  _Float16* kvt    = (_Float16*)carve((size_t)32 * 64 * 64 * 2);
  float*    ksum   = (float*)carve((size_t)32 * 64 * 4);

  // 1) precision conversion (vectorized x4)
  cvt_f32_f16<<<(NTOK / 4 + 255) / 256, 256, 0, stream>>>(x, xh, NTOK / 4);
  cvt_f32_f16<<<(NW / 4 + 255) / 256, 256, 0, stream>>>(wq, wqh, NW / 4);
  cvt_f32_f16<<<(NW / 4 + 255) / 256, 256, 0, stream>>>(wk, wkh, NW / 4);
  cvt_f32_f16<<<(NW / 4 + 255) / 256, 256, 0, stream>>>(wv, wvh, NW / 4);
  cvt_f32_f16<<<(NW / 4 + 255) / 256, 256, 0, stream>>>(wo, woh, NW / 4);

  // 2) Q/K/V projections with fused feature map + head-layout scatter
  dim3 gemm_grid(M / 128, DM / 128);
  gemm_nt_f16<<<gemm_grid, 256, 0, stream>>>(xh, wqh, bq, M, DM, DM, 1, nullptr, qf);
  gemm_nt_f16<<<gemm_grid, 256, 0, stream>>>(xh, wkh, bk, M, DM, DM, 2, nullptr, kft);
  gemm_nt_f16<<<gemm_grid, 256, 0, stream>>>(xh, wvh, bv, M, DM, DM, 3, nullptr, vt);

  // 3) per-head KV (64x64) and ksum
  kv_kernel<<<32, 256, 0, stream>>>(kft, vt, kvt, ksum);

  // 4) numerator GEMM + normalization -> merged
  attn_kernel<<<dim3(16, 32), 256, 0, stream>>>(qf, kvt, ksum, merged);

  // 5) output projection -> f32 d_out
  gemm_nt_f16<<<gemm_grid, 256, 0, stream>>>(merged, woh, bo, M, DM, DM, 0, out, nullptr);
}
